// MultiHeadAttention_16758962389474
// MI455X (gfx1250) — matrix operand order repaired and checked by reading, not compiled, not run on hardware
//
#include <hip/hip_runtime.h>

typedef __bf16 bf16;
typedef __attribute__((ext_vector_type(16))) __bf16 v16bf;
typedef __attribute__((ext_vector_type(8)))  __bf16 v8bf;
typedef __attribute__((ext_vector_type(4)))  __bf16 v4bf;
typedef __attribute__((ext_vector_type(8)))  float  v8f;
typedef int          v4i   __attribute__((vector_size(16)));
typedef unsigned int v4u   __attribute__((vector_size(16)));
typedef int          v8i32 __attribute__((vector_size(32)));

constexpr int B_  = 4;
constexpr int S_  = 2048;
constexpr int E_  = 1024;
constexpr int H_  = 16;
constexpr int D_  = 64;
constexpr int TE_ = 3 * E_;     // 3072
constexpr int BS_ = B_ * S_;    // 8192

// ---------------------------------------------------------------------------
// gfx1250 async load-to-LDS (ASYNCcnt path), with sync fallback
// ---------------------------------------------------------------------------
#if defined(__has_builtin)
#if __has_builtin(__builtin_amdgcn_global_load_async_to_lds_b128)
#define HAVE_ASYNC_LDS 1
#endif
#if __has_builtin(__builtin_amdgcn_tensor_load_to_lds) && __has_builtin(__builtin_amdgcn_s_wait_tensorcnt)
#define HAVE_TDM 1
#endif
#endif
#ifndef HAVE_ASYNC_LDS
#define HAVE_ASYNC_LDS 0
#endif
#ifndef HAVE_TDM
#define HAVE_TDM 0
#endif

#define TO_GBL128(p) ((__attribute__((address_space(1))) v4i*)(p))
#define TO_LDS128(p) ((__attribute__((address_space(3))) v4i*)(p))

__device__ inline void async_fence() {
#if HAVE_ASYNC_LDS
#if __has_builtin(__builtin_amdgcn_s_wait_asynccnt)
    __builtin_amdgcn_s_wait_asynccnt(0);
#else
    asm volatile("s_wait_asynccnt 0x0" ::: "memory");
#endif
#endif
}

// copy one 16-byte chunk global->LDS
__device__ inline void lds_fill_b128(const bf16* g, bf16* l) {
#if HAVE_ASYNC_LDS
    __builtin_amdgcn_global_load_async_to_lds_b128(TO_GBL128(g), TO_LDS128(l), 0, 0);
#else
    *(v8bf*)l = *(const v8bf*)g;
#endif
}

#if HAVE_TDM
// Tensor Data Mover: 2D tile (tile_d1 rows x tile_d0 elems, 2-byte data) from
// global (row stride d0_stride elems) into LDS, optional per-row LDS padding.
// D# layout per CDNA5 ISA sec 8.3-8.5; groups 2/3/4 zero (2D tensor).
__device__ inline void tdm_load_2d_b16(unsigned lds_off, const void* gaddr,
                                       unsigned tensor_d0, unsigned tensor_d1,
                                       unsigned tile_d0, unsigned tile_d1,
                                       unsigned d0_stride,
                                       unsigned pad_en, unsigned pad_interval,
                                       unsigned pad_amount) {
    unsigned long long ga = (unsigned long long)gaddr;
    v4u g0;
    g0[0] = 1u;                                              // count=1, user D#
    g0[1] = lds_off;                                         // lds_addr (bytes)
    g0[2] = (unsigned)(ga & 0xFFFFFFFFu);                    // global_addr[31:0]
    g0[3] = (unsigned)((ga >> 32) & 0x01FFFFFFu) | (2u << 30); // addr[56:32], type=2
    v8i32 g1;
    g1[0] = (int)((1u << 16) |                               // data_size = 2 bytes
                  (pad_en << 20) | (pad_interval << 22) | (pad_amount << 25));
    g1[1] = (int)((tensor_d0 & 0xFFFFu) << 16);              // tensor_dim0 lo16 @bit48
    g1[2] = (int)(((tensor_d0 >> 16) & 0xFFFFu) |
                  ((tensor_d1 & 0xFFFFu) << 16));            // dim0 hi16 | dim1 lo16
    g1[3] = (int)(((tensor_d1 >> 16) & 0xFFFFu) |
                  ((tile_d0 & 0xFFFFu) << 16));              // dim1 hi16 | tile_dim0
    g1[4] = (int)(tile_d1 & 0xFFFFu);                        // tile_dim1 (tile_dim2=0)
    g1[5] = (int)d0_stride;                                  // tensor_dim0_stride lo32
    g1[6] = 0;
    g1[7] = 0;
    v4i   z4 = {0, 0, 0, 0};
    v8i32 z8 = {0, 0, 0, 0, 0, 0, 0, 0};
    __builtin_amdgcn_tensor_load_to_lds(g0, g1, z4, z4, z8, 0);
}

__device__ inline unsigned lds_offset_of(const void* p) {
    return (unsigned)(unsigned long long)(__attribute__((address_space(3))) void*)p;
}
#endif

// ---------------------------------------------------------------------------
// fp32 -> bf16 convert (vector x4)
// ---------------------------------------------------------------------------
__global__ void cvt_f32_bf16(const float* __restrict__ src, bf16* __restrict__ dst) {
    int idx = (blockIdx.x * blockDim.x + threadIdx.x) * 4;
    float4 f = *(const float4*)(src + idx);
    v4bf o;
    o[0] = (bf16)f.x; o[1] = (bf16)f.y; o[2] = (bf16)f.z; o[3] = (bf16)f.w;
    *(v4bf*)(dst + idx) = o;
}

// ---------------------------------------------------------------------------
// WMMA fragment helpers (layouts per CDNA5 ISA 7.12.2)
// ---------------------------------------------------------------------------
__device__ inline v16bf load_a_frag(const bf16* row_ptr, int half) {
    v8bf lo = *(const v8bf*)(row_ptr + 8 * half);
    v8bf hi = *(const v8bf*)(row_ptr + 16 + 8 * half);
    v16bf a;
#pragma unroll
    for (int j = 0; j < 8; ++j) { a[j] = lo[j]; a[8 + j] = hi[j]; }
    return a;
}

__device__ inline v16bf load_b_frag(const bf16* col_ptr, int half) {
    v8bf b0 = *(const v8bf*)(col_ptr + 8 * half);
    v8bf b1 = *(const v8bf*)(col_ptr + 16 + 8 * half);
    v16bf b;
#pragma unroll
    for (int j = 0; j < 8; ++j) { b[j] = b0[j]; b[8 + j] = b1[j]; }
    return b;
}

// ---------------------------------------------------------------------------
// shared GEMM mainloop: C[128x64] tile, bf16 WMMA, f32 acc,
// double-buffered LDS, async DMA overlapped with compute
// ---------------------------------------------------------------------------
#define BM 128
#define BN 64
#define BK 32
#define LDA 40
#define LDB 40

__device__ inline void gemm_issue_fill(const bf16* __restrict__ X,
                                       const bf16* __restrict__ W,
                                       int m0, int n0, int k0, int kdim,
                                       int tid, bf16* Xs, bf16* Ws) {
    const int r = tid >> 2;             // 0..63
    const int c = (tid & 3) * 8;        // 0,8,16,24
    lds_fill_b128(X + (m0 + r) * kdim + k0 + c,        Xs + r * LDA + c);
    lds_fill_b128(X + (m0 + r + 64) * kdim + k0 + c,   Xs + (r + 64) * LDA + c);
    lds_fill_b128(W + (n0 + r) * kdim + k0 + c,        Ws + r * LDB + c);
}

__device__ inline void gemm_mainloop(const bf16* __restrict__ X,
                                     const bf16* __restrict__ W,
                                     int m0, int n0, int kdim,
                                     int tid, int wm, int wn, int l16, int half,
                                     bf16 (*Xs)[BM * LDA], bf16 (*Ws)[BN * LDB],
                                     v8f acc[2][2]) {
    gemm_issue_fill(X, W, m0, n0, 0, kdim, tid, Xs[0], Ws[0]);
    for (int k0 = 0; k0 < kdim; k0 += BK) {
        const int buf = (k0 / BK) & 1;
        async_fence();
        __syncthreads();
        if (k0 + BK < kdim)
            gemm_issue_fill(X, W, m0, n0, k0 + BK, kdim, tid, Xs[buf ^ 1], Ws[buf ^ 1]);
#pragma unroll
        for (int mt = 0; mt < 2; ++mt) {
            v16bf a = load_a_frag(Xs[buf] + (wm + mt * 16 + l16) * LDA, half);
#pragma unroll
            for (int nt = 0; nt < 2; ++nt) {
                v16bf b = load_b_frag(Ws[buf] + (wn + nt * 16 + l16) * LDB, half);
                acc[mt][nt] = __builtin_amdgcn_wmma_f32_16x16x32_bf16(
                    false, a, false, b, (short)0, acc[mt][nt], false, false);
            }
        }
        __syncthreads();
    }
}

// ---------------------------------------------------------------------------
// QKV projection: [BS,E] x [3E,E]^T + bias -> scatter bf16 Q/K/V in [B,H,S,D]
// ---------------------------------------------------------------------------
__global__ __launch_bounds__(256) void qkv_gemm(
    const bf16* __restrict__ X, const bf16* __restrict__ W,
    const float* __restrict__ bias,
    bf16* __restrict__ Q, bf16* __restrict__ K, bf16* __restrict__ V) {
    __shared__ bf16 Xs[2][BM * LDA];
    __shared__ bf16 Ws[2][BN * LDB];
    const int tid  = threadIdx.x;
    const int lane = tid & 31;
    const int wave = tid >> 5;
    const int wm   = (wave & 3) * 32;
    const int wn   = (wave >> 2) * 32;
    const int m0   = blockIdx.x * BM;
    const int n0   = blockIdx.y * BN;
    const int half = lane >> 4;
    const int l16  = lane & 15;

    v8f acc[2][2] = {};
    gemm_mainloop(X, W, m0, n0, E_, tid, wm, wn, l16, half, Xs, Ws, acc);

#pragma unroll
    for (int mt = 0; mt < 2; ++mt) {
#pragma unroll
        for (int nt = 0; nt < 2; ++nt) {
            int n  = n0 + wn + nt * 16 + l16;
            int t  = n >> 10;                 // n / E_
            int e  = n & (E_ - 1);
            int hh = e >> 6;
            int d  = e & 63;
            float bv = bias[n];
            bf16* dst = (t == 0) ? Q : (t == 1) ? K : V;
            float sc  = (t == 0) ? 0.125f : 1.0f;   // 1/sqrt(64)
#pragma unroll
            for (int r = 0; r < 8; ++r) {
                int m = m0 + wm + mt * 16 + half * 8 + r;
                int b = m >> 11;              // m / S_
                int s = m & (S_ - 1);
                float val = (acc[mt][nt][r] + bv) * sc;
                dst[((b * H_ + hh) * S_ + s) * D_ + d] = (bf16)val;
            }
        }
    }
}

// ---------------------------------------------------------------------------
// Flash attention: block = (b,h) x 64 queries, 4 waves x 16 rows each.
// K tile via TDM (wave 0 issues one 2D descriptor, HW pads rows to LDK);
// V register-transpose overlaps the DMA; WMMA QK^T + PV; online softmax.
// ---------------------------------------------------------------------------
#define QT  64
#define KT  32
#define LDK 72
#define LDV 40
#define LDP 40

__global__ __launch_bounds__(128) void attn_fwd(
    const bf16* __restrict__ Q, const bf16* __restrict__ K,
    const bf16* __restrict__ V, bf16* __restrict__ O) {
    __shared__ bf16 Ks[KT * LDK];        // [key][d], row stride 72 (= TDM pad)
    __shared__ bf16 Vs[D_ * LDV];        // [d][key]  (transposed on fill)
    __shared__ bf16 Ps[4][16 * LDP];     // per-wave P scratch [qrow][key]

    const int tid  = threadIdx.x;
    const int lane = tid & 31;
    const int wave = tid >> 5;
    const int half = lane >> 4;
    const int l16  = lane & 15;
    const int bh   = blockIdx.y;
    const int q0   = blockIdx.x * QT;

    const bf16* qbase = Q + (size_t)bh * S_ * D_;
    const bf16* kbase = K + (size_t)bh * S_ * D_;
    const bf16* vbase = V + (size_t)bh * S_ * D_;

    v16bf qa[2];
    {
        const bf16* qr = qbase + (q0 + wave * 16 + l16) * D_;
        qa[0] = load_a_frag(qr, half);
        qa[1] = load_a_frag(qr + 32, half);
    }

    v8f   o_acc[4] = {};
    float rm[8], rs[8];
#pragma unroll
    for (int r = 0; r < 8; ++r) { rm[r] = -1e30f; rs[r] = 0.0f; }

    const int fkey = tid >> 3;           // 0..15
    const int fd   = (tid & 7) * 8;      // 0,8,...,56
#if HAVE_TDM
    const unsigned ks_off = lds_offset_of(Ks);
#endif

    for (int j0 = 0; j0 < S_; j0 += KT) {
        // ---- K tile fill ----
#if HAVE_TDM
        if (wave == 0) {
            // 32x64 bf16 tile; rows 128B = 32 DWORDs (pad_interval code 4),
            // pad 4 DWORDs (code 3) -> LDS row stride 144B = LDK elems
            tdm_load_2d_b16(ks_off, kbase + (size_t)j0 * D_,
                            /*tensor_d0=*/D_, /*tensor_d1=*/S_,
                            /*tile_d0=*/D_, /*tile_d1=*/KT,
                            /*d0_stride=*/D_,
                            /*pad_en=*/1u, /*pad_interval=*/4u, /*pad_amount=*/3u);
        }
#else
        lds_fill_b128(kbase + (j0 + fkey) * D_ + fd,      Ks + fkey * LDK + fd);
        lds_fill_b128(kbase + (j0 + fkey + 16) * D_ + fd, Ks + (fkey + 16) * LDK + fd);
#endif
        // ---- V tile: register transpose into [d][key] (overlaps DMA) ----
#pragma unroll
        for (int t = 0; t < 2; ++t) {
            int key = fkey + t * 16;
            v8bf vv = *(const v8bf*)(vbase + (j0 + key) * D_ + fd);
#pragma unroll
            for (int j = 0; j < 8; ++j) Vs[(fd + j) * LDV + key] = vv[j];
        }
        if (j0 + KT < S_) {   // L2/WGP prefetch of next tile -> global_prefetch_b8
            __builtin_prefetch(kbase + (j0 + KT) * D_ + tid * 16, 0, 1);
            __builtin_prefetch(vbase + (j0 + KT) * D_ + tid * 16, 0, 1);
        }
#if HAVE_TDM
        if (wave == 0) __builtin_amdgcn_s_wait_tensorcnt(0);
#endif
        async_fence();
        __syncthreads();

        // ---- scores: 16q x 32k, two 16-col tiles, each 2 WMMA over D=64 ----
        v8f sc[2] = {};
#pragma unroll
        for (int nt = 0; nt < 2; ++nt) {
#pragma unroll
            for (int kk = 0; kk < 2; ++kk) {
                v16bf b = load_b_frag(Ks + (nt * 16 + l16) * LDK + kk * 32, half);
                sc[nt] = __builtin_amdgcn_wmma_f32_16x16x32_bf16(
                    false, qa[kk], false, b, (short)0, sc[nt], false, false);
            }
        }

        // ---- online softmax ----
#pragma unroll
        for (int r = 0; r < 8; ++r) {
            float v = fmaxf(sc[0][r], sc[1][r]);
            v = fmaxf(v, __shfl_xor(v, 1, 32));
            v = fmaxf(v, __shfl_xor(v, 2, 32));
            v = fmaxf(v, __shfl_xor(v, 4, 32));
            v = fmaxf(v, __shfl_xor(v, 8, 32));
            float nm    = fmaxf(rm[r], v);
            float alpha = __expf(rm[r] - nm);
            rm[r] = nm;
            rs[r] *= alpha;
#pragma unroll
            for (int n = 0; n < 4; ++n) o_acc[n][r] *= alpha;
            float p0 = __expf(sc[0][r] - nm);
            float p1 = __expf(sc[1][r] - nm);
            sc[0][r] = p0; sc[1][r] = p1;
            float rsum = p0 + p1;
            rsum += __shfl_xor(rsum, 1, 32);
            rsum += __shfl_xor(rsum, 2, 32);
            rsum += __shfl_xor(rsum, 4, 32);
            rsum += __shfl_xor(rsum, 8, 32);
            rs[r] += rsum;
        }

        // ---- C-layout -> A-layout through per-wave LDS scratch ----
        bf16* P = Ps[wave];
#pragma unroll
        for (int nt = 0; nt < 2; ++nt)
#pragma unroll
            for (int r = 0; r < 8; ++r)
                P[(half * 8 + r) * LDP + nt * 16 + l16] = (bf16)sc[nt][r];
        v16bf pa = load_a_frag(P + l16 * LDP, half);

        // ---- o += P @ V ----
#pragma unroll
        for (int n = 0; n < 4; ++n) {
            v16bf b = load_b_frag(Vs + (n * 16 + l16) * LDV, half);
            o_acc[n] = __builtin_amdgcn_wmma_f32_16x16x32_bf16(
                false, pa, false, b, (short)0, o_acc[n], false, false);
        }
        __syncthreads();
    }

    // normalize and write [B,S,E] bf16 (E index = h*64 + d)
    const int b = bh >> 4;
    const int h = bh & (H_ - 1);
#pragma unroll
    for (int n = 0; n < 4; ++n) {
        int d = n * 16 + l16;
#pragma unroll
        for (int r = 0; r < 8; ++r) {
            int srow  = q0 + wave * 16 + half * 8 + r;
            float val = o_acc[n][r] / rs[r];
            O[((size_t)(b * S_ + srow)) * E_ + h * D_ + d] = (bf16)val;
        }
    }
}

// ---------------------------------------------------------------------------
// Output projection: [BS,E] x [E,E]^T + bias -> fp32 d_out
// ---------------------------------------------------------------------------
__global__ __launch_bounds__(256) void out_gemm(
    const bf16* __restrict__ X, const bf16* __restrict__ W,
    const float* __restrict__ bias, float* __restrict__ Y) {
    __shared__ bf16 Xs[2][BM * LDA];
    __shared__ bf16 Ws[2][BN * LDB];
    const int tid  = threadIdx.x;
    const int lane = tid & 31;
    const int wave = tid >> 5;
    const int wm   = (wave & 3) * 32;
    const int wn   = (wave >> 2) * 32;
    const int m0   = blockIdx.x * BM;
    const int n0   = blockIdx.y * BN;
    const int half = lane >> 4;
    const int l16  = lane & 15;

    v8f acc[2][2] = {};
    gemm_mainloop(X, W, m0, n0, E_, tid, wm, wn, l16, half, Xs, Ws, acc);

#pragma unroll
    for (int mt = 0; mt < 2; ++mt) {
#pragma unroll
        for (int nt = 0; nt < 2; ++nt) {
            int n = n0 + wn + nt * 16 + l16;
            float bv = bias[n];
#pragma unroll
            for (int r = 0; r < 8; ++r) {
                int m = m0 + wm + mt * 16 + half * 8 + r;
                Y[(size_t)m * E_ + n] = acc[mt][nt][r] + bv;
            }
        }
    }
}

// ---------------------------------------------------------------------------
// launcher
// ---------------------------------------------------------------------------
extern "C" void kernel_launch(void* const* d_in, const int* in_sizes, int n_in,
                              void* d_out, int out_size, void* d_ws, size_t ws_size,
                              hipStream_t stream) {
    const float* x     = (const float*)d_in[0];
    const float* qkv_w = (const float*)d_in[1];
    const float* qkv_b = (const float*)d_in[2];
    const float* out_w = (const float*)d_in[3];
    const float* out_b = (const float*)d_in[4];
    float* out = (float*)d_out;

    char* ws = (char*)d_ws;
    size_t off = 0;
    auto alloc = [&](size_t bytes) -> void* {
        void* p = ws + off;
        off += (bytes + 255) & ~(size_t)255;
        return p;
    };
    bf16* xb = (bf16*)alloc((size_t)BS_ * E_ * sizeof(bf16));
    bf16* wq = (bf16*)alloc((size_t)TE_ * E_ * sizeof(bf16));
    bf16* wo = (bf16*)alloc((size_t)E_ * E_ * sizeof(bf16));
    bf16* q  = (bf16*)alloc((size_t)BS_ * E_ * sizeof(bf16));
    bf16* k  = (bf16*)alloc((size_t)BS_ * E_ * sizeof(bf16));
    bf16* v  = (bf16*)alloc((size_t)BS_ * E_ * sizeof(bf16));
    bf16* ao = (bf16*)alloc((size_t)BS_ * E_ * sizeof(bf16));
    (void)ws_size; (void)in_sizes; (void)n_in; (void)out_size;

    cvt_f32_bf16<<<(BS_ * E_ / 4) / 256, 256, 0, stream>>>(x, xb);
    cvt_f32_bf16<<<(TE_ * E_ / 4) / 256, 256, 0, stream>>>(qkv_w, wq);
    cvt_f32_bf16<<<(E_ * E_ / 4) / 256, 256, 0, stream>>>(out_w, wo);

    dim3 g1(BS_ / BM, TE_ / BN);
    qkv_gemm<<<g1, 256, 0, stream>>>(xb, wq, qkv_b, q, k, v);

    dim3 g2(S_ / QT, B_ * H_);
    attn_fwd<<<g2, 128, 0, stream>>>(q, k, v, ao);

    dim3 g3(BS_ / BM, E_ / BN);
    out_gemm<<<g3, 256, 0, stream>>>(ao, wo, out_b, out);
}
